// Model_16604343566947
// MI455X (gfx1250) — matrix operand order repaired and checked by reading, not compiled, not run on hardware
//
#include <hip/hip_runtime.h>
#include <hip/hip_bf16.h>

// ---------------------------------------------------------------------------
// emb-gather -> LSTM(128, seq) -> LSTM(128, last) -> Dense64+ReLU -> Dense1+sig
// B=256 T=512 V=32000 E=H=128.
//
// MI455X plan:
//  * Recurrence independent per batch row -> 16 chunks of 16 rows per layer.
//  * BOTH layers run concurrently in one 32-WG kernel: layer-2 WGs trail
//    layer-1 WGs by 2 timesteps via agent-scope acquire/release progress
//    flags. The per-layer scan loops are SEPARATE template instantiations
//    (branch hoisted outside the loop) so each gets clean register
//    allocation -- no cross-path phi copies dragging s_wait_loadcnt up
//    before the WMMA chain.
//  * Per step: all gates via one [16x256]@[256x512] bf16 WMMA GEMM with
//    register-resident weights; clamped-index software-pipelined x prefetch;
//    DScnt-only barriers keep LOADcnt in flight across sync points.
// ---------------------------------------------------------------------------

#define BB   256
#define TT   512
#define EE   128
#define HH   128
#define G4   512     // 4*H gate width
#define KK   256     // E + H contraction
#define MROWS 16
#define NCHUNK (BB / MROWS)   // 16

typedef __attribute__((ext_vector_type(16))) __bf16 v16bf;
typedef __attribute__((ext_vector_type(8)))  __bf16 v8bf;
typedef __attribute__((ext_vector_type(4)))  __bf16 v4bf;
typedef __attribute__((ext_vector_type(2)))  __bf16 v2bf;
typedef __attribute__((ext_vector_type(8)))  float  v8f;
typedef __attribute__((ext_vector_type(4)))  float  v4f;
typedef __attribute__((ext_vector_type(2)))  float  v2f;
typedef __attribute__((ext_vector_type(4)))  unsigned v4u;

// Workgroup barrier draining only LDS (DScnt); global loads/stores stay live.
__device__ __forceinline__ void barrier_lds() {
    asm volatile("s_wait_dscnt 0x0\n\t"
                 "s_barrier_signal -1\n\t"
                 "s_barrier_wait -1" ::: "memory");
}
// Drain this thread's outstanding global stores (ack from L2).
__device__ __forceinline__ void drain_stores() {
    asm volatile("s_wait_storecnt 0x0" ::: "memory");
}

__device__ __forceinline__ float sigf(float x) {
    return __builtin_amdgcn_rcpf(1.0f + __expf(-x));   // branch-free
}
__device__ __forceinline__ float tanh_fast(float x) {
    float e = __expf(2.0f * x);                        // inf-safe, no NaN
    return 1.0f - 2.0f * __builtin_amdgcn_rcpf(e + 1.0f);
}

// All waves poll: each wave's acquire invalidates its own WGP$ path.
__device__ __forceinline__ void wait_prog(const int* p, int needed) {
    while (__hip_atomic_load(p, __ATOMIC_ACQUIRE, __HIP_MEMORY_SCOPE_AGENT) < needed)
        __builtin_amdgcn_s_sleep(2);
}

// ---------------------------------------------------------------------------
// WUt[n][k] = (k<128 ? W[k][n] : U[k-128][n]) as bf16; WUt: [512 x 256].
// Also re-zeros the progress flags (required for deterministic graph replay).
// ---------------------------------------------------------------------------
__global__ void prep_weights(const float* __restrict__ W1, const float* __restrict__ U1,
                             const float* __restrict__ W2, const float* __restrict__ U2,
                             __bf16* __restrict__ wu1t, __bf16* __restrict__ wu2t,
                             int* __restrict__ prog)
{
    int idx = blockIdx.x * blockDim.x + threadIdx.x;   // 0 .. 512*256-1
    if (idx < NCHUNK) prog[idx] = 0;
    int n = idx >> 8;
    int k = idx & 255;
    float a = (k < EE) ? W1[(size_t)k * G4 + n] : U1[(size_t)(k - EE) * G4 + n];
    float b = (k < EE) ? W2[(size_t)k * G4 + n] : U2[(size_t)(k - EE) * G4 + n];
    wu1t[idx] = (__bf16)a;
    wu2t[idx] = (__bf16)b;
}

// ---------------------------------------------------------------------------
// One layer's full scan. FIRST: x = emb[texts], publishes h1seq + pflag.
//                       !FIRST: x = h1seq gated on pflag, writes final h.
// ---------------------------------------------------------------------------
template<bool FIRST>
__device__ __forceinline__ void scan_body(
    __bf16 (&xbuf)[2][MROWS][EE],
    __bf16 (&hbuf)[MROWS][HH],
    float  (&zbuf)[MROWS][G4],
    int    (&toks)[MROWS][TT],
    const int*    __restrict__ texts,
    const float*  __restrict__ emb,
    const __bf16* __restrict__ wut,      // [512][256] bf16
    const float*  __restrict__ bias,     // [512]
    __bf16*       __restrict__ h1seq,    // [B][T][H] bf16 (read xor written)
    float*        __restrict__ hlast,    // [B][H] f32 (layer 2 out)
    int*          __restrict__ pflag,    // this chunk's progress flag
    int b0)
{
    const int tid  = threadIdx.x;
    const int wave = tid >> 5;
    const int lane = tid & 31;

    // ---- init: h=0, tokens to LDS (layer 1 only) ----
    for (int e = tid; e < MROWS * HH; e += 1024) ((__bf16*)hbuf)[e] = (__bf16)0.0f;
    if (FIRST) {
        for (int e = tid; e < MROWS * TT / 4; e += 1024)
            *(v4u*)&((int*)toks)[e * 4] = *(const v4u*)&texts[(size_t)b0 * TT + e * 4];
    }

    // ---- register-resident B fragments (contiguous k order, same as A:
    //      lane n<16 = col n, K 0..15; lane n+16 = col n, K 16..31) ----
    const int n0 = wave * 16;
    const int l  = lane & 15;
    const int bko   = (lane < 16) ? 0 : 16;
    const int akoff = (lane < 16) ? 0 : 16;
    v16bf bfrag[8];
    {
        const __bf16* rowp = wut + (size_t)(n0 + l) * KK;
#pragma unroll
        for (int kt = 0; kt < 8; ++kt)
            bfrag[kt] = *(const v16bf*)(rowp + kt * 32 + bko);
    }

    // ---- per-thread gate-phase constants & state (2 adjacent columns) ----
    const int gm = tid >> 6;            // batch row 0..15
    const int gj = (tid & 63) * 2;      // h column (even) 0..126
    v2f biasI = *(const v2f*)&bias[gj];
    v2f biasF = *(const v2f*)&bias[EE + gj];
    v2f biasG = *(const v2f*)&bias[2 * EE + gj];
    v2f biasO = *(const v2f*)&bias[3 * EE + gj];
    v2f creg = {0.0f, 0.0f};            // cell state lives in registers

    // ---- stage x for t=0 ----
    if (FIRST) {
        if (tid < 512) {
            int m = tid >> 5, c4 = (tid & 31) * 4;
            int tok = texts[(size_t)(b0 + m) * TT];
            v4f x = *(const v4f*)&emb[(size_t)tok * EE + c4];
            v4bf p = {(__bf16)x[0], (__bf16)x[1], (__bf16)x[2], (__bf16)x[3]};
            *(v4bf*)&xbuf[0][m][c4] = p;
        }
    } else {
        wait_prog(pflag, 1);            // h1[t=0] published?
        if (tid < 256) {
            int m = tid >> 4, c8 = (tid & 15) * 8;
            *(v4u*)&xbuf[0][m][c8] =
                *(const v4u*)&h1seq[((size_t)(b0 + m) * TT) * HH + c8];
        }
    }
    barrier_lds();

    // ================= sequential scan =================
    for (int t = 0; t < TT; ++t) {
        const int cur = t & 1, nxt = cur ^ 1;
        // Clamped prefetch step: unconditional issue/retire each iteration.
        const int tp = (t + 1 < TT) ? (t + 1) : (TT - 1);

        // ---- issue prefetch loads for step tp (stay in flight across the zbuf
        //      barrier: barrier_lds does not drain LOADcnt) ----
        v4f preX;
        v4u preH;
        if (FIRST) {
            if (tid < 512) {
                int m = tid >> 5, c4 = (tid & 31) * 4;
                int tok = toks[m][tp];
                preX = *(const v4f*)&emb[(size_t)tok * EE + c4];
            }
        } else {
            const int needed = (t + 2 < TT) ? (t + 2) : TT;
            wait_prog(pflag, needed);       // h1[tp] published?
            if (tid < 256) {
                int m = tid >> 4, c8 = (tid & 15) * 8;
                preH = *(const v4u*)&h1seq[((size_t)(b0 + m) * TT + tp) * HH + c8];
            }
        }

        // ---- A fragments + WMMA chain ----
        // A (16x32), contiguous k order as B: lane l<16 = row l {K0..15}; l+16 {K16..31}
        v16bf afrag[8];
#pragma unroll
        for (int kt = 0; kt < 8; ++kt) {
            const __bf16* ap = (kt < 4) ? &xbuf[cur][l][kt * 32 + akoff]
                                        : &hbuf[l][(kt - 4) * 32 + akoff];
            v8bf lo = *(const v8bf*)(ap);
            v8bf hi = *(const v8bf*)(ap + 8);
            afrag[kt] = __builtin_shufflevector(lo, hi,
                0, 1, 2, 3, 4, 5, 6, 7, 8, 9, 10, 11, 12, 13, 14, 15);
        }
        v8f acc = {};
#pragma unroll
        for (int kt = 0; kt < 8; ++kt)
            acc = __builtin_amdgcn_wmma_f32_16x16x32_bf16(
                false, afrag[kt], false, bfrag[kt], (short)0, acc, false, false);

        // C/D layout: lane = N + 16*(M>=8); vgpr v = M%8
        {
            int n = lane & 15, mb = (lane >> 4) * 8;
#pragma unroll
            for (int v = 0; v < 8; ++v) zbuf[mb + v][n0 + n] = acc[v];
        }
        barrier_lds();

        // ---- gate math: 2 adjacent columns per thread, branch-free ----
        {
            v2f zi = *(const v2f*)&zbuf[gm][gj]          + biasI;
            v2f zf = *(const v2f*)&zbuf[gm][EE + gj]     + biasF;
            v2f zg = *(const v2f*)&zbuf[gm][2 * EE + gj] + biasG;
            v2f zo = *(const v2f*)&zbuf[gm][3 * EE + gj] + biasO;
            float hh[2];
#pragma unroll
            for (int q = 0; q < 2; ++q) {
                float ig = sigf(zi[q]);
                float fg = sigf(zf[q]);
                float gg = tanh_fast(zg[q]);
                float og = sigf(zo[q]);
                float c  = fg * creg[q] + ig * gg;
                creg[q] = c;
                hh[q] = og * tanh_fast(c);
            }
            v2bf hp = {(__bf16)hh[0], (__bf16)hh[1]};
            *(v2bf*)&hbuf[gm][gj] = hp;
            if (FIRST) {
                *(v2bf*)&h1seq[((size_t)(b0 + gm) * TT + t) * HH + gj] = hp;
            } else if (t == TT - 1) {
                *(v2f*)&hlast[(size_t)(b0 + gm) * HH + gj] = (v2f){hh[0], hh[1]};
            }
        }

        // ---- retire prefetch into the other x buffer ----
        if (FIRST) {
            if (tid < 512) {
                int m = tid >> 5, c4 = (tid & 31) * 4;
                v4bf p = {(__bf16)preX[0], (__bf16)preX[1],
                          (__bf16)preX[2], (__bf16)preX[3]};
                *(v4bf*)&xbuf[nxt][m][c4] = p;
            }
            // publish h1[t]: own stores ack'd in L2, then one release store
            drain_stores();
            barrier_lds();
            if (tid == 0)
                __hip_atomic_store(pflag, t + 1, __ATOMIC_RELEASE,
                                   __HIP_MEMORY_SCOPE_AGENT);
        } else {
            if (tid < 256) {
                int m = tid >> 4, c8 = (tid & 15) * 8;
                *(v4u*)&xbuf[nxt][m][c8] = preH;
            }
            barrier_lds();
        }
    }
}

// ---------------------------------------------------------------------------
// Blocks 0..15: layer 1. Blocks 16..31: layer 2 (trails via prog flags).
// One WG per WGP; 32 WGPs active, layers overlapped.
// ---------------------------------------------------------------------------
__global__ __launch_bounds__(1024)
void lstm_pipelined(const int*    __restrict__ texts,
                    const float*  __restrict__ emb,
                    const __bf16* __restrict__ wu1t,
                    const __bf16* __restrict__ wu2t,
                    const float*  __restrict__ bias1,
                    const float*  __restrict__ bias2,
                    __bf16*       __restrict__ h1seq,   // [B][T][H] bf16
                    float*        __restrict__ hlast,   // [B][H] f32
                    int*          __restrict__ prog)    // [NCHUNK]
{
    __shared__ __bf16 xbuf[2][MROWS][EE];   // double-buffered x_t   (8 KB)
    __shared__ __bf16 hbuf[MROWS][HH];      // h_{t-1}               (4 KB)
    __shared__ float  zbuf[MROWS][G4];      // gate pre-activations  (32 KB)
    __shared__ int    toks[MROWS][TT];      // token ids (layer 1)   (32 KB)

    if (blockIdx.x < NCHUNK) {
        const int chunk = blockIdx.x;
        scan_body<true>(xbuf, hbuf, zbuf, toks, texts, emb, wu1t, bias1,
                        h1seq, nullptr, prog + chunk, chunk * MROWS);
    } else {
        const int chunk = blockIdx.x - NCHUNK;
        scan_body<false>(xbuf, hbuf, zbuf, toks, texts, emb, wu2t, bias2,
                         h1seq, hlast, prog + chunk, chunk * MROWS);
    }
}

// ---------------------------------------------------------------------------
__global__ void dense_head(const float* __restrict__ h2,
                           const float* __restrict__ Wd, const float* __restrict__ bd,
                           const float* __restrict__ Wo, const float* __restrict__ bo,
                           float* __restrict__ out)
{
    int b = blockIdx.x * blockDim.x + threadIdx.x;
    if (b >= BB) return;
    float d[64];
#pragma unroll
    for (int j = 0; j < 64; ++j) d[j] = bd[j];
    for (int k = 0; k < HH; ++k) {
        float hv = h2[(size_t)b * HH + k];
#pragma unroll
        for (int j = 0; j < 64; ++j) d[j] = fmaf(hv, Wd[(size_t)k * 64 + j], d[j]);
    }
    float acc = bo[0];
#pragma unroll
    for (int j = 0; j < 64; ++j) acc = fmaf(fmaxf(d[j], 0.0f), Wo[j], acc);
    out[b] = sigf(acc);
}

// ---------------------------------------------------------------------------
extern "C" void kernel_launch(void* const* d_in, const int* in_sizes, int n_in,
                              void* d_out, int out_size, void* d_ws, size_t ws_size,
                              hipStream_t stream)
{
    const int*   texts = (const int*)  d_in[0];
    const float* emb   = (const float*)d_in[1];
    const float* W1    = (const float*)d_in[2];
    const float* U1    = (const float*)d_in[3];
    const float* b1    = (const float*)d_in[4];
    const float* W2    = (const float*)d_in[5];
    const float* U2    = (const float*)d_in[6];
    const float* b2    = (const float*)d_in[7];
    const float* Wd    = (const float*)d_in[8];
    const float* bd    = (const float*)d_in[9];
    const float* Wo    = (const float*)d_in[10];
    const float* bo    = (const float*)d_in[11];
    float* out = (float*)d_out;

    __bf16* wu1t  = (__bf16*)d_ws;                          // 512*256 bf16
    __bf16* wu2t  = wu1t + (size_t)G4 * KK;                 // 512*256 bf16
    __bf16* h1seq = wu2t + (size_t)G4 * KK;                 // B*T*H bf16 (33.5 MB)
    float*  h2    = (float*)(h1seq + (size_t)BB * TT * HH); // B*H f32
    int*    prog  = (int*)(h2 + (size_t)BB * HH);           // NCHUNK flags

    prep_weights<<<G4 * KK / 256, 256, 0, stream>>>(W1, U1, W2, U2,
                                                    wu1t, wu2t, prog);
    lstm_pipelined<<<2 * NCHUNK, 1024, 0, stream>>>(texts, emb, wu1t, wu2t,
                                                    b1, b2, h1seq, h2, prog);
    dense_head<<<1, 256, 0, stream>>>(h2, Wd, bd, Wo, bo, out);
}
